// GNN_82351702933810
// MI455X (gfx1250) — matrix operand order repaired and checked by reading, not compiled, not run on hardware
//
#include <hip/hip_runtime.h>

// ---------------- problem constants ----------------
constexpr int N_NODES   = 100000;
constexpr int N_EDGES   = 3200000;
constexpr int N_FEAT    = 165;     // K of layer-1 GEMM
constexpr int HIDDEN    = 495;     // N of layer-1 GEMM
constexpr int N_CLASSES = 2;

constexpr int KP1  = 192;          // N_FEAT padded to 6 k-tiles of 32 (also agg1 row stride)
constexpr int KT1  = 6;            // k-tiles, layer 1
constexpr int NT1  = 31;           // n-tiles of 16 covering HIDDEN (496)
constexpr int H2P  = 512;          // HIDDEN padded for layer-2 K
constexpr int KT2  = 16;           // k-tiles, layer 2
constexpr int NBLK = N_NODES / 16; // 6250 row blocks (exact)

// ---------------- WMMA types ----------------
typedef __attribute__((ext_vector_type(16))) __bf16 v16bf;
typedef __attribute__((ext_vector_type(8)))  float  v8f;

struct alignas(16) U4 { unsigned int a, b, c, d; };
union V16BF { v16bf v; U4 q[2]; unsigned short u[16]; };

__device__ __forceinline__ unsigned short bf16_bits(float f) {
    unsigned u = __builtin_bit_cast(unsigned, f);
    unsigned r = u + 0x7FFFu + ((u >> 16) & 1u);   // round-to-nearest-even
    return (unsigned short)(r >> 16);
}

// gfx1250 async global->LDS path (ASYNCcnt), guarded so absence can't break the build.
#if defined(__gfx1250__) && __has_builtin(__builtin_amdgcn_global_load_async_to_lds_b128) && \
    __has_builtin(__builtin_amdgcn_s_wait_asynccnt)
#define USE_ASYNC_LDS 1
#else
#define USE_ASYNC_LDS 0
#endif

// The b128 async builtin takes pointers to 4xi32 vectors in AS1 (global) / AS3 (LDS).
typedef int v4i_vs __attribute__((vector_size(16)));
// generic->AS3: low 32 bits of a generic LDS pointer are the ds byte offset (flat aperture rule)
#define AS1V4(p) ((__attribute__((address_space(1))) v4i_vs*)(unsigned long long)(p))
#define AS3V4(p) ((__attribute__((address_space(3))) v4i_vs*)(unsigned int)(unsigned long long)(p))

// ---------------- 1) init: deg=1 (self loop), zero accumulators ----------------
__global__ void k_init(float* deg, float* agg1, float* agg2) {
    size_t i = (size_t)blockIdx.x * blockDim.x + threadIdx.x;
    if (i < (size_t)N_NODES * KP1) agg1[i] = 0.0f;      // zero incl. K-pad columns
    if (i < (size_t)N_NODES)       deg[i]  = 1.0f;      // self-loop contribution
    if (i < (size_t)(2 * N_NODES)) agg2[i] = 0.0f;
}

// ---------------- 2) degree via f32 atomics (L2-resident) ----------------
__global__ void k_degree(const long long* __restrict__ ei, float* __restrict__ deg) {
    int i = blockIdx.x * blockDim.x + threadIdx.x;
    if (i >= N_EDGES) return;
    atomicAdd(&deg[(int)ei[N_EDGES + i]], 1.0f);
}

// ---------------- 3) deg -> deg^-1/2 in place ----------------
__global__ void k_dinv(float* deg) {
    int i = blockIdx.x * blockDim.x + threadIdx.x;
    if (i < N_NODES) deg[i] = rsqrtf(deg[i]);           // deg >= 1 always
}

// ---------------- 4) agg1[d] += w * x[s], 165-wide, one wave32 per edge ----------------
__global__ void k_agg_feat(const float* __restrict__ x, const long long* __restrict__ ei,
                           const float* __restrict__ dinv, float* __restrict__ agg1) {
    int gtid = blockIdx.x * blockDim.x + threadIdx.x;
    int e    = gtid >> 5;
    int lane = gtid & 31;
    if (e >= N_EDGES + N_NODES) return;
    int s, d;
    if (e < N_EDGES) { s = (int)ei[e]; d = (int)ei[N_EDGES + e]; }
    else             { s = d = e - N_EDGES; }           // self loop
    float w = dinv[s] * dinv[d];
    const float* xs = x    + (size_t)s * N_FEAT;
    float*       ad = agg1 + (size_t)d * KP1;           // padded row stride
    for (int f = lane; f < N_FEAT; f += 32)
        atomicAdd(&ad[f], w * xs[f]);                   // x+agg1 fit in 192MB L2
}

// ---------------- 5) pack W1 -> bf16 in per-lane WMMA B order ----------------
// idx = ((nt*KT1+kt)*32 + lane)*16 + j ; lane: col = lane&15 ; K = kt*32 + (j>>3)*16 + (lane>>4)*8 + (j&7)
__global__ void k_packW1(const float* __restrict__ W1, unsigned short* __restrict__ W1p) {
    int idx = blockIdx.x * blockDim.x + threadIdx.x;
    if (idx >= NT1 * KT1 * 512) return;
    int j    = idx & 15;
    int lane = (idx >> 4) & 31;
    int t    = idx >> 9;
    int kt   = t % KT1, nt = t / KT1;
    int n    = nt * 16 + (lane & 15);
    int k    = kt * 32 + (j >> 3) * 16 + (lane >> 4) * 8 + (j & 7);
    float v  = (k < N_FEAT && n < HIDDEN) ? W1[k * HIDDEN + n] : 0.0f;
    W1p[idx] = bf16_bits(v);
}

// ---------------- 6) pack W2 (K padded to 512, N padded to 16) ----------------
__global__ void k_packW2(const float* __restrict__ W2, unsigned short* __restrict__ W2p) {
    int idx = blockIdx.x * blockDim.x + threadIdx.x;
    if (idx >= KT2 * 512) return;
    int j    = idx & 15;
    int lane = (idx >> 4) & 31;
    int kt   = idx >> 9;
    int n    = lane & 15;
    int k    = kt * 32 + (j >> 3) * 16 + (lane >> 4) * 8 + (j & 7);
    float v  = (k < HIDDEN && n < N_CLASSES) ? W2[k * N_CLASSES + n] : 0.0f;
    W2p[idx] = bf16_bits(v);
}

// ---------------- 7) fused GEMM1 + bias + ReLU + GEMM2, h1 lives in LDS ----------------
__global__ __launch_bounds__(256) void k_gemm_fused(
    const float* __restrict__ agg1, const unsigned short* __restrict__ W1p,
    const float* __restrict__ b1,   const unsigned short* __restrict__ W2p,
    float* __restrict__ h2) {
    __shared__ float          stage[16][KP1];  // fp32 A tile (12 KB), async-filled
    __shared__ unsigned short h1buf[16][H2P];  // bf16 relu(h1) tile (16 KB)

    const int tid  = threadIdx.x;
    const int lane = tid & 31;
    const int wave = tid >> 5;
    const int row  = lane & 15;     // M index of A fragment / C column
    const int hf   = lane >> 4;     // K-half selector (A) / M-half selector (C)
    const size_t mbase = (size_t)blockIdx.x * 16;

    // stage A tile: one contiguous 12288B region (16 rows x 192 f32, 16B aligned)
    const float* gsrc = agg1 + mbase * KP1;
#if USE_ASYNC_LDS
    // 768 b128 transfers = exactly 3 rounds of 256 lanes, DMA'd without touching VGPRs
#pragma unroll
    for (int it = 0; it < 3; ++it) {
        int i = it * 256 + tid;                 // chunk index, 16B each
        __builtin_amdgcn_global_load_async_to_lds_b128(
            AS1V4(gsrc + i * 4), AS3V4(&stage[0][0] + i * 4), 0, 0);
    }
#else
    {
        float* st = &stage[0][0];
        for (int i = tid; i < 16 * KP1; i += 256) st[i] = gsrc[i];
    }
#endif
    for (int i = tid; i < 16 * H2P; i += 256) (&h1buf[0][0])[i] = 0;
#if USE_ASYNC_LDS
    __builtin_amdgcn_s_wait_asynccnt(0);
#endif
    __syncthreads();

    // build all 6 bf16 A fragments per wave (16B-aligned ds_load_b128 + cvt)
    V16BF afrag[KT1];
#pragma unroll
    for (int kt = 0; kt < KT1; ++kt) {
        int k0 = kt * 32 + hf * 8;
        float4 f0 = *(const float4*)&stage[row][k0];
        float4 f1 = *(const float4*)&stage[row][k0 + 4];
        float4 f2 = *(const float4*)&stage[row][k0 + 16];
        float4 f3 = *(const float4*)&stage[row][k0 + 20];
        afrag[kt].u[0]  = bf16_bits(f0.x); afrag[kt].u[1]  = bf16_bits(f0.y);
        afrag[kt].u[2]  = bf16_bits(f0.z); afrag[kt].u[3]  = bf16_bits(f0.w);
        afrag[kt].u[4]  = bf16_bits(f1.x); afrag[kt].u[5]  = bf16_bits(f1.y);
        afrag[kt].u[6]  = bf16_bits(f1.z); afrag[kt].u[7]  = bf16_bits(f1.w);
        afrag[kt].u[8]  = bf16_bits(f2.x); afrag[kt].u[9]  = bf16_bits(f2.y);
        afrag[kt].u[10] = bf16_bits(f2.z); afrag[kt].u[11] = bf16_bits(f2.w);
        afrag[kt].u[12] = bf16_bits(f3.x); afrag[kt].u[13] = bf16_bits(f3.y);
        afrag[kt].u[14] = bf16_bits(f3.z); afrag[kt].u[15] = bf16_bits(f3.w);
    }

    // layer-1: 31 n-tiles split across 8 waves, K reduction via WMMA
    const v16bf* Bp = (const v16bf*)W1p;
    for (int nt = wave; nt < NT1; nt += 8) {
        v8f acc = {};
#pragma unroll
        for (int kt = 0; kt < KT1; ++kt) {
            v16bf b = Bp[(nt * KT1 + kt) * 32 + lane];
            acc = __builtin_amdgcn_wmma_f32_16x16x32_bf16(
                false, afrag[kt].v, false, b, (short)0, acc, false, false);
        }
        int col = nt * 16 + (lane & 15);
        if (col < HIDDEN) {
            float bias = b1[col];
#pragma unroll
            for (int r = 0; r < 8; ++r) {
                float v = fmaxf(acc[r] + bias, 0.0f);      // + b1, ReLU
                h1buf[r + hf * 8][col] = bf16_bits(v);     // C layout: M = r + 8*hf
            }
        }
    }
    __syncthreads();

    // layer-2: h2 = relu(h1) @ W2 (K=512 padded, only cols 0..1 valid)
    if (wave == 0) {
        const v16bf* B2p = (const v16bf*)W2p;
        v8f acc = {};
#pragma unroll
        for (int kt = 0; kt < KT2; ++kt) {
            V16BF a;
            int k0 = kt * 32 + hf * 8;
            a.q[0] = *(const U4*)&h1buf[row][k0];
            a.q[1] = *(const U4*)&h1buf[row][k0 + 16];
            v16bf b = B2p[kt * 32 + lane];
            acc = __builtin_amdgcn_wmma_f32_16x16x32_bf16(
                false, a.v, false, b, (short)0, acc, false, false);
        }
        int col = lane & 15;
        if (col < N_CLASSES) {
#pragma unroll
            for (int r = 0; r < 8; ++r)
                h2[(mbase + r + hf * 8) * 2 + col] = acc[r];
        }
    }
}

// ---------------- 8) 2-wide aggregation of h2 ----------------
__global__ void k_agg_h2(const float* __restrict__ h2, const long long* __restrict__ ei,
                         const float* __restrict__ dinv, float* __restrict__ agg2) {
    int e = blockIdx.x * blockDim.x + threadIdx.x;
    if (e >= N_EDGES + N_NODES) return;
    int s, d;
    if (e < N_EDGES) { s = (int)ei[e]; d = (int)ei[N_EDGES + e]; }
    else             { s = d = e - N_EDGES; }
    float w = dinv[s] * dinv[d];
    atomicAdd(&agg2[2 * d + 0], w * h2[2 * s + 0]);
    atomicAdd(&agg2[2 * d + 1], w * h2[2 * s + 1]);
}

// ---------------- 9) + b2, log_softmax over 2 classes ----------------
__global__ void k_logsoftmax(const float* __restrict__ agg2, const float* __restrict__ b2,
                             float* __restrict__ out) {
    int i = blockIdx.x * blockDim.x + threadIdx.x;
    if (i >= N_NODES) return;
    float a0 = agg2[2 * i + 0] + b2[0];
    float a1 = agg2[2 * i + 1] + b2[1];
    float m  = fmaxf(a0, a1);
    float lse = m + logf(expf(a0 - m) + expf(a1 - m));
    out[2 * i + 0] = a0 - lse;
    out[2 * i + 1] = a1 - lse;
}

// ---------------- host launcher ----------------
extern "C" void kernel_launch(void* const* d_in, const int* in_sizes, int n_in,
                              void* d_out, int out_size, void* d_ws, size_t ws_size,
                              hipStream_t stream) {
    const float*     x  = (const float*)d_in[0];
    const long long* ei = (const long long*)d_in[1];   // int64 [2, E]
    const float*     W1 = (const float*)d_in[2];
    const float*     b1 = (const float*)d_in[3];
    const float*     W2 = (const float*)d_in[4];
    const float*     b2 = (const float*)d_in[5];
    float*           out = (float*)d_out;

    char* ws = (char*)d_ws;
    auto carve = [&](size_t bytes) {
        char* p = ws;
        ws += (bytes + 255) & ~(size_t)255;
        return p;
    };
    float*          deg  = (float*)carve((size_t)N_NODES * 4);            // deg -> dinv
    float*          agg1 = (float*)carve((size_t)N_NODES * KP1 * 4);      // 76.8 MB, K-padded
    float*          h2   = (float*)carve((size_t)N_NODES * 2 * 4);
    float*          agg2 = (float*)carve((size_t)N_NODES * 2 * 4);
    unsigned short* W1p  = (unsigned short*)carve((size_t)NT1 * KT1 * 512 * 2);
    unsigned short* W2p  = (unsigned short*)carve((size_t)KT2 * 512 * 2);

    const int T = 256;
    size_t initN = (size_t)N_NODES * KP1;
    k_init      <<<(unsigned)((initN + T - 1) / T), T, 0, stream>>>(deg, agg1, agg2);
    k_degree    <<<(N_EDGES + T - 1) / T, T, 0, stream>>>(ei, deg);
    k_dinv      <<<(N_NODES + T - 1) / T, T, 0, stream>>>(deg);
    {   // one wave32 per (edge + self-loop)
        int waves  = N_EDGES + N_NODES;
        int blocks = (waves + 7) / 8;                  // 8 waves per 256-thread block
        k_agg_feat<<<blocks, T, 0, stream>>>(x, ei, deg, agg1);
    }
    k_packW1    <<<(NT1 * KT1 * 512 + T - 1) / T, T, 0, stream>>>(W1, W1p);
    k_packW2    <<<(KT2 * 512 + T - 1) / T, T, 0, stream>>>(W2, W2p);
    k_gemm_fused<<<NBLK, T, 0, stream>>>(agg1, W1p, b1, W2p, h2);
    k_agg_h2    <<<(N_EDGES + N_NODES + T - 1) / T, T, 0, stream>>>(h2, ei, deg, agg2);
    k_logsoftmax<<<(N_NODES + T - 1) / T, T, 0, stream>>>(agg2, b2, out);
}
